// Model_88064009437895
// MI455X (gfx1250) — matrix operand order repaired and checked by reading, not compiled, not run on hardware
//
#include <hip/hip_runtime.h>
#include <stdint.h>
#include <stddef.h>

#define N_NODES    50000
#define N_EDGES    800000
#define N_NONEDGES 400000
#define N_GRAPHS   1000
#define EMB        64
#define NBLK       6
#define GEN_EPS    1e-7f
#define ATT_SCALE  0.125f            // 1/sqrt(64)
#define ENC_NEG_INF 0x007FFFFFu      // enc(-inf)

typedef _Float16 v16h __attribute__((ext_vector_type(16)));
typedef _Float16 v8h  __attribute__((ext_vector_type(8)));
typedef float    v8f  __attribute__((ext_vector_type(8)));

// order-preserving float <-> uint encoding for atomicMax on floats
__device__ __forceinline__ unsigned enc_f32(float f) {
  unsigned b = __float_as_uint(f);
  return (b & 0x80000000u) ? ~b : (b | 0x80000000u);
}
__device__ __forceinline__ float dec_f32(unsigned u) {
  unsigned b = (u & 0x80000000u) ? (u & 0x7FFFFFFFu) : ~u;
  return __uint_as_float(b);
}

__device__ __forceinline__ float leaky01(float v) {
  return v > 0.0f ? v : 0.01f * v;
}

__global__ void k_fill_u32(unsigned* __restrict__ p, unsigned v, long n) {
  long i = (long)blockIdx.x * blockDim.x + threadIdx.x;
  if (i < n) p[i] = v;
}

// Transpose + f32->f16 convert all 64x64 weights once (41 matrices).
__global__ void k_wt_convert(const float* tq, const float* tk, const float* tv,
                             const float* te, const float* ts, const float* tg,
                             const float* h0, const float* h1, const float* h2,
                             const float* h3, const float* h4,
                             _Float16* __restrict__ wt) {
  int b = blockIdx.x;
  const float* src;
  if (b < 36) {
    int fam = b / 6, i = b % 6;
    const float* f = fam == 0 ? tq : fam == 1 ? tk : fam == 2 ? tv
                   : fam == 3 ? te : fam == 4 ? ts : tg;
    src = f + (size_t)i * 4096;
  } else {
    int h = b - 36;
    src = h == 0 ? h0 : h == 1 ? h1 : h == 2 ? h2 : h == 3 ? h3 : h4;
  }
  _Float16* dst = wt + (size_t)b * 4096;
  for (int idx = threadIdx.x; idx < 4096; idx += blockDim.x) {
    int kk = idx >> 6, n = idx & 63;
    dst[n * 64 + kk] = (_Float16)src[kk * 64 + n];   // N-major for WMMA B frags
  }
}

// C[M,64] = act( A[M,64] @ W[64,64] + bias ), A optionally gathered as
// A[idx0[r*s]] + A[idx1[r*s]].  W given f16, N-major (Wt[n*64+k]).
// Block: 256 threads = 8 waves, tile 64 rows x 64 cols.  Each wave owns two
// 16x16 M-tiles sharing one B fragment -> 4x v_wmma_f32_16x16x32_f16.
// B (weights) staged to LDS with GLOBAL_LOAD_ASYNC_TO_LDS_B128 (ASYNCcnt).
__global__ __launch_bounds__(256)
void k_gemm64(const float* __restrict__ A,
              const int* __restrict__ idx0, const int* __restrict__ idx1,
              int idxStride,
              const _Float16* __restrict__ Wt, const float* __restrict__ bias,
              float* __restrict__ C, int M, int act) {
  __shared__ _Float16 As[64 * 80];   // padded stride 80 halves (160B)
  __shared__ _Float16 Bs[64 * 80];
  int t = threadIdx.x;

  // stage B: 64x64 halves; async DMA straight to LDS, 2x16B per thread
  {
    int n  = t >> 2;
    int ko = (t & 3) * 16;
    unsigned lo = (unsigned)(uintptr_t)(&Bs[n * 80 + ko]);
    unsigned go = (unsigned)((n * 64 + ko) * 2);
    asm volatile("global_load_async_to_lds_b128 %0, %1, %2 offset:0"
                 :: "v"(lo), "v"(go), "s"(Wt) : "memory");
    asm volatile("global_load_async_to_lds_b128 %0, %1, %2 offset:16"
                 :: "v"(lo + 16), "v"(go), "s"(Wt) : "memory");
  }
  // stage A: 64 rows, f32 -> f16 convert (optional 2-way gather-sum)
  {
    int r  = t >> 2;
    int co = (t & 3) * 16;
    long rG = (long)blockIdx.x * 64 + r;
    _Float16 hv[16];
    if (rG < M) {
      if (idx0) {
        long ia = idx0[rG * idxStride];
        long ib = idx1[rG * idxStride];
        const float* pa = A + ia * 64 + co;
        const float* pb = A + ib * 64 + co;
#pragma unroll
        for (int j = 0; j < 16; ++j) hv[j] = (_Float16)(pa[j] + pb[j]);
      } else {
        const float* pa = A + rG * 64 + co;
#pragma unroll
        for (int j = 0; j < 16; ++j) hv[j] = (_Float16)pa[j];
      }
    } else {
#pragma unroll
      for (int j = 0; j < 16; ++j) hv[j] = (_Float16)0.0f;
    }
    float4* d = reinterpret_cast<float4*>(&As[r * 80 + co]);
    d[0] = *reinterpret_cast<float4*>(&hv[0]);
    d[1] = *reinterpret_cast<float4*>(&hv[8]);
  }
  asm volatile("s_wait_asynccnt 0x0" ::: "memory");
  __syncthreads();

  int lane = t & 31;
  int wv   = t >> 5;
  int tm = (wv >> 2) * 16;           // 0,16  (second M-tile at +32)
  int tn = (wv & 3) * 16;            // 0..48
  int mr = lane & 15;
  int ka = (lane >> 4) * 8;          // A frag: K {ka..ka+7, 16+ka..16+ka+7}
  int kb = (lane >> 4) * 8;          // B frag: K {kb..kb+7, 16+kb..16+kb+7}
  v8f acc0 = {};
  v8f acc1 = {};
#pragma unroll
  for (int ks = 0; ks < 64; ks += 32) {
    const _Float16* bp = &Bs[(tn + mr) * 80 + ks + kb];
    v8h blo = *reinterpret_cast<const v8h*>(bp);
    v8h bhi = *reinterpret_cast<const v8h*>(bp + 16);
    v16h bf = __builtin_shufflevector(blo, bhi,
        0, 1, 2, 3, 4, 5, 6, 7, 8, 9, 10, 11, 12, 13, 14, 15);
    const _Float16* ap0 = &As[(tm + mr) * 80 + ks + ka];
    v8h a0l = *reinterpret_cast<const v8h*>(ap0);
    v8h a0h = *reinterpret_cast<const v8h*>(ap0 + 16);
    v16h a0 = __builtin_shufflevector(a0l, a0h,
        0, 1, 2, 3, 4, 5, 6, 7, 8, 9, 10, 11, 12, 13, 14, 15);
    const _Float16* ap1 = &As[(tm + 32 + mr) * 80 + ks + ka];
    v8h a1l = *reinterpret_cast<const v8h*>(ap1);
    v8h a1h = *reinterpret_cast<const v8h*>(ap1 + 16);
    v16h a1 = __builtin_shufflevector(a1l, a1h,
        0, 1, 2, 3, 4, 5, 6, 7, 8, 9, 10, 11, 12, 13, 14, 15);
    acc0 = __builtin_amdgcn_wmma_f32_16x16x32_f16(
        false, a0, false, bf, (short)0, acc0, false, false);
    acc1 = __builtin_amdgcn_wmma_f32_16x16x32_f16(
        false, a1, false, bf, (short)0, acc1, false, false);
  }
  int col = tn + (lane & 15);
  float bv = bias ? bias[col] : 0.0f;
  long row0 = (long)blockIdx.x * 64 + tm + (lane >> 4) * 8;
  float* cp = C + row0 * 64 + col;
  if ((long)blockIdx.x * 64 + 64 <= M) {        // full tile: branchless stores
#pragma unroll
    for (int d = 0; d < 8; ++d) {
      float v0 = acc0[d] + bv;
      float v1 = acc1[d] + bv;
      if (act) { v0 = leaky01(v0); v1 = leaky01(v1); }
      cp[(long)d * 64]        = v0;
      cp[(long)(d + 32) * 64] = v1;
    }
  } else {
#pragma unroll
    for (int d = 0; d < 8; ++d) {
      if (row0 + d < M) {
        float v0 = acc0[d] + bv;
        if (act) v0 = leaky01(v0);
        cp[(long)d * 64] = v0;
      }
      if (row0 + 32 + d < M) {
        float v1 = acc1[d] + bv;
        if (act) v1 = leaky01(v1);
        cp[(long)(d + 32) * 64] = v1;
      }
    }
  }
}

__global__ void k_node_embed(const float* __restrict__ x, const float* __restrict__ W,
                             const float* __restrict__ b, float* __restrict__ o) {
  __shared__ float xs[4][16];
  int ln = threadIdx.x >> 6;
  int c  = threadIdx.x & 63;
  long n = (long)blockIdx.x * 4 + ln;
  if (n < N_NODES && c < 16) xs[ln][c] = x[n * 16 + c];
  __syncthreads();
  if (n >= N_NODES) return;
  float acc = b[c];
#pragma unroll
  for (int kk = 0; kk < 16; ++kk) acc += xs[ln][kk] * W[kk * 64 + c];
  o[n * 64 + c] = acc;
}

__global__ void k_edge_embed(const float* __restrict__ ea, const float* __restrict__ W,
                             const float* __restrict__ b, float* __restrict__ e) {
  long tid = (long)blockIdx.x * blockDim.x + threadIdx.x;
  if (tid >= (long)N_EDGES * 64) return;
  long m = tid >> 6;
  int c = tid & 63;
  float acc = b[c];
#pragma unroll
  for (int kk = 0; kk < 4; ++kk) acc += ea[m * 4 + kk] * W[kk * 64 + c];
  e[tid] = acc;
}

// wave-per-edge: alpha_raw = scale * sum_c q[dst]*(k[src]+ee)
__global__ void k_edge_alpha(const float* __restrict__ q, const float* __restrict__ k,
                             const float* __restrict__ ee, const int* __restrict__ ei,
                             float* __restrict__ alpha) {
  long e = (long)blockIdx.x * 8 + (threadIdx.x >> 5);
  int lane = threadIdx.x & 31;
  if (e >= N_EDGES) return;
  long src = ei[e], dst = ei[N_EDGES + e];
  float s = q[dst * 64 + lane]      * (k[src * 64 + lane]      + ee[e * 64 + lane]) +
            q[dst * 64 + lane + 32] * (k[src * 64 + lane + 32] + ee[e * 64 + lane + 32]);
#pragma unroll
  for (int off = 16; off > 0; off >>= 1) s += __shfl_xor(s, off, 32);
  if (lane == 0) alpha[e] = s * ATT_SCALE;
}

__global__ void k_seg_max(const float* __restrict__ alpha, const int* __restrict__ ei,
                          unsigned* __restrict__ segmax) {
  long e = (long)blockIdx.x * blockDim.x + threadIdx.x;
  if (e >= N_EDGES) return;
  atomicMax(&segmax[ei[N_EDGES + e]], enc_f32(alpha[e]));
}

__global__ void k_seg_exp(float* __restrict__ alpha, const int* __restrict__ ei,
                          const unsigned* __restrict__ segmax, float* __restrict__ segsum) {
  long e = (long)blockIdx.x * blockDim.x + threadIdx.x;
  if (e >= N_EDGES) return;
  int dst = ei[N_EDGES + e];
  float ex = __expf(alpha[e] - dec_f32(segmax[dst]));
  alpha[e] = ex;
  atomicAdd(&segsum[dst], ex);
}

// agg[dst] += (v[src]+ee) * alpha/(sum+eps)   (wave-per-edge, 2 ch/lane)
__global__ void k_agg(const float* __restrict__ v, const float* __restrict__ ee,
                      const float* __restrict__ alpha, const float* __restrict__ segsum,
                      const int* __restrict__ ei, float* __restrict__ agg) {
  long e = (long)blockIdx.x * 8 + (threadIdx.x >> 5);
  int lane = threadIdx.x & 31;
  if (e >= N_EDGES) return;
  long src = ei[e], dst = ei[N_EDGES + e];
  float a = alpha[e] / (segsum[dst] + 1e-16f);
  atomicAdd(&agg[dst * 64 + lane],      (v[src * 64 + lane]      + ee[e * 64 + lane]) * a);
  atomicAdd(&agg[dst * 64 + lane + 32], (v[src * 64 + lane + 32] + ee[e * 64 + lane + 32]) * a);
}

__global__ void k_node_upd(const float* __restrict__ agg, const float* __restrict__ skip,
                           float* __restrict__ o, float* __restrict__ tmp) {
  long i = (long)blockIdx.x * blockDim.x + threadIdx.x;
  if (i >= (long)N_NODES * 64) return;
  float v = agg[i] + skip[i];
  o[i] = v;
  tmp[i] = v;
}

// tmp[dst] += relu(o[src]+e) + eps
__global__ void k_msg(const float* __restrict__ o, const float* __restrict__ e,
                      const int* __restrict__ ei, float* __restrict__ tmp) {
  long ed = (long)blockIdx.x * 8 + (threadIdx.x >> 5);
  int lane = threadIdx.x & 31;
  if (ed >= N_EDGES) return;
  long src = ei[ed], dst = ei[N_EDGES + ed];
  float m0 = o[src * 64 + lane]      + e[ed * 64 + lane];
  float m1 = o[src * 64 + lane + 32] + e[ed * 64 + lane + 32];
  m0 = (m0 > 0.0f ? m0 : 0.0f) + GEN_EPS;
  m1 = (m1 > 0.0f ? m1 : 0.0f) + GEN_EPS;
  atomicAdd(&tmp[dst * 64 + lane], m0);
  atomicAdd(&tmp[dst * 64 + lane + 32], m1);
}

__global__ void k_pool(const float* __restrict__ o, const int* __restrict__ batch,
                       float* __restrict__ glob, float* __restrict__ counts) {
  long i = (long)blockIdx.x * blockDim.x + threadIdx.x;
  if (i >= (long)N_NODES * 64) return;
  long n = i >> 6;
  int c = i & 63;
  int b = batch[n];
  atomicAdd(&glob[(long)b * 64 + c], o[i]);
  if (c == 0) atomicAdd(&counts[b], 1.0f);
}

__global__ void k_glob_div(float* __restrict__ glob, const float* __restrict__ counts) {
  long i = (long)blockIdx.x * blockDim.x + threadIdx.x;
  if (i >= (long)N_GRAPHS * 64) return;
  glob[i] = glob[i] / fmaxf(counts[i >> 6], 1.0f);
}

// second (narrow) layer of each MLP head: out = hid @ W2 + b2, dout<=8
__global__ void k_head2(const float* __restrict__ hid, const float* __restrict__ W2,
                        const float* __restrict__ b2, float* __restrict__ out,
                        long M, int dout) {
  long tid = (long)blockIdx.x * blockDim.x + threadIdx.x;
  if (tid >= M * dout) return;
  long r = tid / dout;
  int j = (int)(tid - r * dout);
  float acc = b2[j];
#pragma unroll 16
  for (int c = 0; c < 64; ++c) acc += hid[r * 64 + c] * W2[c * dout + j];
  out[tid] = acc;
}

extern "C" void kernel_launch(void* const* d_in, const int* in_sizes, int n_in,
                              void* d_out, int out_size, void* d_ws, size_t ws_size,
                              hipStream_t stream) {
  (void)in_sizes; (void)n_in; (void)out_size; (void)ws_size;
  // ---- inputs (setup_inputs dict order, params flattened in insertion order)
  const float* x      = (const float*)d_in[0];
  const float* ea     = (const float*)d_in[1];
  const int*   ei     = (const int*)d_in[2];
  const int*   nei    = (const int*)d_in[3];
  const int*   batch  = (const int*)d_in[4];
  const float* x2h_W  = (const float*)d_in[5];
  const float* x2h_b  = (const float*)d_in[6];
  const float* e2h_W  = (const float*)d_in[7];
  const float* e2h_b  = (const float*)d_in[8];
  const float* tcq_W  = (const float*)d_in[9];
  const float* tcq_b  = (const float*)d_in[10];
  const float* tck_W  = (const float*)d_in[11];
  const float* tck_b  = (const float*)d_in[12];
  const float* tcv_W  = (const float*)d_in[13];
  const float* tcv_b  = (const float*)d_in[14];
  const float* tce_W  = (const float*)d_in[15];
  /* d_in[16] = tc_e zero bias (unused) */
  const float* tcs_W  = (const float*)d_in[17];
  const float* tcs_b  = (const float*)d_in[18];
  const float* gen_W  = (const float*)d_in[19];
  const float* gen_b  = (const float*)d_in[20];
  const float* stop_W1 = (const float*)d_in[21];
  const float* stop_b1 = (const float*)d_in[22];
  const float* stop_W2 = (const float*)d_in[23];
  const float* stop_b2 = (const float*)d_in[24];
  const float* an_W1 = (const float*)d_in[25];
  const float* an_b1 = (const float*)d_in[26];
  const float* an_W2 = (const float*)d_in[27];
  const float* an_b2 = (const float*)d_in[28];
  const float* ae_W1 = (const float*)d_in[29];
  const float* ae_b1 = (const float*)d_in[30];
  const float* ae_W2 = (const float*)d_in[31];
  const float* ae_b2 = (const float*)d_in[32];
  const float* se_W1 = (const float*)d_in[33];
  const float* se_b1 = (const float*)d_in[34];
  const float* se_W2 = (const float*)d_in[35];
  const float* se_b2 = (const float*)d_in[36];
  const float* rw_W1 = (const float*)d_in[37];
  const float* rw_b1 = (const float*)d_in[38];
  const float* rw_W2 = (const float*)d_in[39];
  const float* rw_b2 = (const float*)d_in[40];
  float* out = (float*)d_out;

  // ---- workspace carve-up (256B aligned)
  char* wsb = (char*)d_ws;
  size_t off = 0;
  auto take = [&](size_t bytes) -> char* {
    char* p = wsb + off;
    off += (bytes + 255) & ~(size_t)255;
    return p;
  };
  float* o      = (float*)take((size_t)N_NODES * 64 * 4);
  float* q      = (float*)take((size_t)N_NODES * 64 * 4);
  float* kbuf   = (float*)take((size_t)N_NODES * 64 * 4);
  float* v      = (float*)take((size_t)N_NODES * 64 * 4);
  float* skip   = (float*)take((size_t)N_NODES * 64 * 4);
  float* tmp    = (float*)take((size_t)N_NODES * 64 * 4);
  float* agg    = (float*)take((size_t)N_NODES * 64 * 4);
  float* ebuf   = (float*)take((size_t)N_EDGES * 64 * 4);
  float* ee     = (float*)take((size_t)N_EDGES * 64 * 4);  // reused as head hidden
  float* alpha  = (float*)take((size_t)N_EDGES * 4);
  unsigned* segmax = (unsigned*)take((size_t)N_NODES * 4);
  float* segsum = (float*)take((size_t)N_NODES * 4);
  float* glob   = (float*)take((size_t)N_GRAPHS * 64 * 4);
  float* counts = (float*)take((size_t)N_GRAPHS * 4);
  _Float16* wt  = (_Float16*)take((size_t)41 * 4096 * 2);

  const int T = 256;
  // weights -> f16, transposed (N-major)
  k_wt_convert<<<41, T, 0, stream>>>(tcq_W, tck_W, tcv_W, tce_W, tcs_W, gen_W,
                                     stop_W1, an_W1, ae_W1, se_W1, rw_W1, wt);
  // embeddings
  k_node_embed<<<(N_NODES + 3) / 4, T, 0, stream>>>(x, x2h_W, x2h_b, o);
  k_edge_embed<<<(int)(((long)N_EDGES * 64 + T - 1) / T), T, 0, stream>>>(ea, e2h_W, e2h_b, ebuf);

  const int gbN = (N_NODES + 63) / 64;
  const int gbE = (N_EDGES + 63) / 64;
  for (int i = 0; i < NBLK; ++i) {
    const _Float16* Wq = wt + (size_t)(0 * 6 + i) * 4096;
    const _Float16* Wk = wt + (size_t)(1 * 6 + i) * 4096;
    const _Float16* Wv = wt + (size_t)(2 * 6 + i) * 4096;
    const _Float16* We = wt + (size_t)(3 * 6 + i) * 4096;
    const _Float16* Ws = wt + (size_t)(4 * 6 + i) * 4096;
    const _Float16* Wm = wt + (size_t)(5 * 6 + i) * 4096;

    k_gemm64<<<gbE, T, 0, stream>>>(ebuf, nullptr, nullptr, 0, We, nullptr, ee, N_EDGES, 0);
    k_gemm64<<<gbN, T, 0, stream>>>(o, nullptr, nullptr, 0, Wq, tcq_b + i * 64, q,    N_NODES, 0);
    k_gemm64<<<gbN, T, 0, stream>>>(o, nullptr, nullptr, 0, Wk, tck_b + i * 64, kbuf, N_NODES, 0);
    k_gemm64<<<gbN, T, 0, stream>>>(o, nullptr, nullptr, 0, Wv, tcv_b + i * 64, v,    N_NODES, 0);
    k_gemm64<<<gbN, T, 0, stream>>>(o, nullptr, nullptr, 0, Ws, tcs_b + i * 64, skip, N_NODES, 0);

    k_edge_alpha<<<(N_EDGES + 7) / 8, T, 0, stream>>>(q, kbuf, ee, ei, alpha);
    k_fill_u32<<<(N_NODES + T - 1) / T, T, 0, stream>>>(segmax, ENC_NEG_INF, N_NODES);
    k_fill_u32<<<(N_NODES + T - 1) / T, T, 0, stream>>>((unsigned*)segsum, 0u, N_NODES);
    k_seg_max<<<(N_EDGES + T - 1) / T, T, 0, stream>>>(alpha, ei, segmax);
    k_seg_exp<<<(N_EDGES + T - 1) / T, T, 0, stream>>>(alpha, ei, segmax, segsum);
    k_fill_u32<<<(int)(((long)N_NODES * 64 + T - 1) / T), T, 0, stream>>>((unsigned*)agg, 0u, (long)N_NODES * 64);
    k_agg<<<(N_EDGES + 7) / 8, T, 0, stream>>>(v, ee, alpha, segsum, ei, agg);
    k_node_upd<<<(int)(((long)N_NODES * 64 + T - 1) / T), T, 0, stream>>>(agg, skip, o, tmp);
    k_msg<<<(N_EDGES + 7) / 8, T, 0, stream>>>(o, ebuf, ei, tmp);
    k_gemm64<<<gbN, T, 0, stream>>>(tmp, nullptr, nullptr, 0, Wm, gen_b + i * 64, o, N_NODES, 0);
  }

  // global mean pool
  k_fill_u32<<<(N_GRAPHS * 64 + T - 1) / T, T, 0, stream>>>((unsigned*)glob, 0u, (long)N_GRAPHS * 64);
  k_fill_u32<<<(N_GRAPHS + T - 1) / T, T, 0, stream>>>((unsigned*)counts, 0u, N_GRAPHS);
  k_pool<<<(int)(((long)N_NODES * 64 + T - 1) / T), T, 0, stream>>>(o, batch, glob, counts);
  k_glob_div<<<(N_GRAPHS * 64 + T - 1) / T, T, 0, stream>>>(glob, counts);

  // heads: hidden GEMM (leaky-relu epilogue, gathers fused) + narrow layer
  float* hid = ee;
  const int gbG = (N_GRAPHS + 63) / 64;
  const int gbH = (N_NONEDGES + 63) / 64;  // 400000 rows for both edge heads

  // stop: glob -> 1000x1
  k_gemm64<<<gbG, T, 0, stream>>>(glob, nullptr, nullptr, 0, wt + (size_t)36 * 4096, stop_b1, hid, N_GRAPHS, 1);
  k_head2<<<(N_GRAPHS + T - 1) / T, T, 0, stream>>>(hid, stop_W2, stop_b2, out + 0, N_GRAPHS, 1);
  // add_node: o -> 50000x8
  k_gemm64<<<gbN, T, 0, stream>>>(o, nullptr, nullptr, 0, wt + (size_t)37 * 4096, an_b1, hid, N_NODES, 1);
  k_head2<<<(int)(((long)N_NODES * 8 + T - 1) / T), T, 0, stream>>>(hid, an_W2, an_b2, out + 1000, N_NODES, 8);
  // add_edge: o[ne_row]+o[ne_col] -> 400000x1
  k_gemm64<<<gbH, T, 0, stream>>>(o, nei, nei + N_NONEDGES, 1, wt + (size_t)38 * 4096, ae_b1, hid, N_NONEDGES, 1);
  k_head2<<<(N_NONEDGES + T - 1) / T, T, 0, stream>>>(hid, ae_W2, ae_b2, out + 401000, N_NONEDGES, 1);
  // set_edge_attr: o[ei[0,::2]]+o[ei[1,::2]] -> 400000x8
  k_gemm64<<<gbH, T, 0, stream>>>(o, ei, ei + N_EDGES, 2, wt + (size_t)39 * 4096, se_b1, hid, N_NONEDGES, 1);
  k_head2<<<(int)(((long)N_NONEDGES * 8 + T - 1) / T), T, 0, stream>>>(hid, se_W2, se_b2, out + 801000, N_NONEDGES, 8);
  // reward: glob -> 1000x1
  k_gemm64<<<gbG, T, 0, stream>>>(glob, nullptr, nullptr, 0, wt + (size_t)40 * 4096, rw_b1, hid, N_GRAPHS, 1);
  k_head2<<<(N_GRAPHS + T - 1) / T, T, 0, stream>>>(hid, rw_W2, rw_b2, out + 4001000, N_GRAPHS, 1);
}
